// MyModel_8778913153293
// MI455X (gfx1250) — matrix operand order repaired and checked by reading, not compiled, not run on hardware
//
#include <hip/hip_runtime.h>

#define NCLS 64     // N
#define KSHOT 64    // K
#define DDIM 640    // D
#define QROWS 131072
#define INV_TAU 10.0f

typedef __attribute__((ext_vector_type(8)))  float  v8f;
typedef __attribute__((ext_vector_type(16))) __bf16 v16bf;
typedef __attribute__((ext_vector_type(4)))  int    v4i;

#define GLOBAL_AS __attribute__((address_space(1)))
#define LDS_AS    __attribute__((address_space(3)))

__device__ __forceinline__ unsigned short f2bf_rne(float x) {
    unsigned u = __float_as_uint(x);
    u += 0x7FFFu + ((u >> 16) & 1u);          // round-to-nearest-even
    return (unsigned short)(u >> 16);
}
__device__ __forceinline__ float bf2f(unsigned short b) {
    return __uint_as_float((unsigned)b << 16);
}

__device__ __forceinline__ float block_reduce_sum(float v, volatile float* red) {
    #pragma unroll
    for (int o = 16; o > 0; o >>= 1) v += __shfl_xor(v, o, 32);
    __syncthreads();
    if ((threadIdx.x & 31) == 0) red[threadIdx.x >> 5] = v;
    __syncthreads();
    float r = (threadIdx.x < 8) ? red[threadIdx.x] : 0.0f;
    if (threadIdx.x < 32) {
        #pragma unroll
        for (int o = 4; o > 0; o >>= 1) r += __shfl_xor(r, o, 32);
        if (threadIdx.x == 0) red[0] = r;
    }
    __syncthreads();
    return red[0];
}

// ---------------------------------------------------------------------------
// Kernel 1: one block per class i. Computes prototypes[i][:] (fp32 math exactly
// as reference), then writes bf16 hi/lo pre-swizzled into WMMA B-fragment
// layout:  flatIdx = (((kb*4 + t)*32) + lane)*16 + e
// where B[k][n] = proto[n][k], n = t*16 + (lane&15), k = kb*32 + (lane>>4)*16 + e
// ---------------------------------------------------------------------------
__global__ __launch_bounds__(256)
void proto_kernel(const float* __restrict__ emb,
                  unsigned short* __restrict__ Bhi,
                  unsigned short* __restrict__ Blo) {
    const int i   = blockIdx.x;
    const int tid = threadIdx.x;
    __shared__ float pu[DDIM];     // proto_unnorm, later reused for proto_dis
    __shared__ float dis[NCLS];
    __shared__ float red[8];

    // proto_unnorm[i][d] = mean_a emb[a*64 + i][d]
    for (int d = tid; d < DDIM; d += 256) {
        float s = 0.0f;
        #pragma unroll 4
        for (int a = 0; a < KSHOT; ++a) s += emb[(size_t)(a * NCLS + i) * DDIM + d];
        pu[d] = s * (1.0f / (float)KSHOT);
    }
    __syncthreads();

    // p_norm
    float ps = 0.0f;
    for (int d = tid; d < DDIM; d += 256) { float v = pu[d]; ps += v * v; }
    float pn = fmaxf(sqrtf(block_reduce_sum(ps, red)), 1e-8f);

    // dis[j] = exp( (pu . row_j) / (pn * ||row_j||) ), row_j = emb[i*64 + j]
    {
        const int g = tid >> 2, l4 = tid & 3;          // 64 groups of 4 lanes
        const float* row = emb + (size_t)(i * NCLS + g) * DDIM;
        float num = 0.0f, s2 = 0.0f;
        for (int d = l4; d < DDIM; d += 4) {
            float x = row[d];
            num += pu[d] * x;
            s2  += x * x;
        }
        num += __shfl_xor(num, 1, 32); num += __shfl_xor(num, 2, 32);
        s2  += __shfl_xor(s2, 1, 32);  s2  += __shfl_xor(s2, 2, 32);
        if (l4 == 0) {
            float sn = fmaxf(sqrtf(s2), 1e-8f);
            dis[g] = expf(num / (pn * sn));
        }
    }
    __syncthreads();

    float invSum = 1.0f / block_reduce_sum((tid < NCLS) ? dis[tid] : 0.0f, red);

    // proto_dis[d] = sum_j w[j] * emb[i*64+j][d]  (overwrite pu)
    float np = 0.0f;
    for (int d = tid; d < DDIM; d += 256) {
        float s = 0.0f;
        #pragma unroll 4
        for (int j = 0; j < NCLS; ++j) s += dis[j] * emb[(size_t)(i * NCLS + j) * DDIM + d];
        s *= invSum;
        pu[d] = s;
        np += s * s;
    }
    float invN = 1.0f / fmaxf(sqrtf(block_reduce_sum(np, red)), 1e-12f);

    // write swizzled bf16 hi/lo B fragments (RNE split; runs once, keep accurate)
    const int t = i >> 4;
    const int lanebase = i & 15;
    for (int d = tid; d < DDIM; d += 256) {
        float p = pu[d] * invN;
        int kb = d >> 5, kk = d & 31, half = kk >> 4, e = kk & 15;
        int lane = lanebase | (half << 4);
        int idx  = (((kb * 4 + t) * 32) + lane) * 16 + e;
        unsigned short hi = f2bf_rne(p);
        Bhi[idx] = hi;
        Blo[idx] = f2bf_rne(p - bf2f(hi));
    }
}

// ---------------------------------------------------------------------------
// Kernel 2: logits = query @ proto^T / TAU via v_wmma_f32_16x16x32_bf16.
// Block = 8 waves x 16 rows = 128 query rows. B (hi+lo, 160 KB) staged in LDS
// via async global->LDS copy. bf16x3 with fp32 accumulate:
//   acc += a_lo*b_hi + a_hi*b_lo + a_hi*b_hi
// A split per k-step: hi = trunc-to-bf16 (exact mask), lo = x - hi_f (trunc),
// packed 2-at-a-time with v_perm_b32 to keep VALU under the WMMA shadow.
// ---------------------------------------------------------------------------
__global__ __launch_bounds__(256)
void logits_kernel(const float* __restrict__ query,
                   const unsigned short* __restrict__ Bgl,   // Bhi | Blo contiguous
                   float* __restrict__ out) {
    extern __shared__ unsigned short sB[];   // 81920 ushorts = 160 KB (hi | lo)

    // ---- stage 160 KB of swizzled B into LDS ----
#if defined(__gfx1250__) && __has_builtin(__builtin_amdgcn_global_load_async_to_lds_b128)
    {
        GLOBAL_AS v4i* g = (GLOBAL_AS v4i*)const_cast<unsigned short*>(Bgl);
        LDS_AS    v4i* s = (LDS_AS v4i*)sB;
        for (int idx = threadIdx.x; idx < 10240; idx += 256) {
            __builtin_amdgcn_global_load_async_to_lds_b128(g + idx, s + idx, 0, 0);
        }
    }
  #if __has_builtin(__builtin_amdgcn_s_wait_asynccnt)
    __builtin_amdgcn_s_wait_asynccnt(0);
  #else
    asm volatile("s_wait_asynccnt 0x0" ::: "memory");
  #endif
#else
    {
        const uint4* g = reinterpret_cast<const uint4*>(Bgl);
        uint4* s = reinterpret_cast<uint4*>(sB);
        for (int idx = threadIdx.x; idx < 10240; idx += 256) s[idx] = g[idx];
    }
#endif
    __syncthreads();

    const int lane = threadIdx.x & 31;
    const int wv   = threadIdx.x >> 5;
    const int half = lane >> 4;
    const int l15  = lane & 15;
    const size_t m0 = (size_t)blockIdx.x * 128 + (size_t)wv * 16;

    const float* arow = query + (m0 + l15) * DDIM + half * 16;
    const v16bf* sBh = reinterpret_cast<const v16bf*>(sB);
    const v16bf* sBl = sBh + 2560;           // 40960/16

    v8f acc[4];
    #pragma unroll
    for (int t = 0; t < 4; ++t) acc[t] = v8f{};

    union Frag { v16bf v; unsigned u[8]; };

    for (int kb = 0; kb < 20; ++kb) {
        // A fragment: 16x32 bf16 layout. This lane needs K = kb*32 + half*16 + [0..15]
        // of its row -> four float4 loads.
        const float* p = arow + kb * 32;
        float xv[16];
        *reinterpret_cast<float4*>(&xv[0])  = *reinterpret_cast<const float4*>(p);
        *reinterpret_cast<float4*>(&xv[4])  = *reinterpret_cast<const float4*>(p + 4);
        *reinterpret_cast<float4*>(&xv[8])  = *reinterpret_cast<const float4*>(p + 8);
        *reinterpret_cast<float4*>(&xv[12]) = *reinterpret_cast<const float4*>(p + 12);

        Frag ah, al;
        #pragma unroll
        for (int j = 0; j < 8; ++j) {
            unsigned u0 = __float_as_uint(xv[2 * j]);
            unsigned u1 = __float_as_uint(xv[2 * j + 1]);
            // pack the two high halves (truncated bf16) in one v_perm_b32
            ah.u[j] = __builtin_amdgcn_perm(u1, u0, 0x07060302u);
            float l0 = xv[2 * j]     - __uint_as_float(u0 & 0xFFFF0000u);
            float l1 = xv[2 * j + 1] - __uint_as_float(u1 & 0xFFFF0000u);
            al.u[j] = __builtin_amdgcn_perm(__float_as_uint(l1), __float_as_uint(l0),
                                            0x07060302u);
        }

        const int bbase = kb * 4 * 32 + lane;
        #pragma unroll
        for (int t = 0; t < 4; ++t) {
            v16bf bh = sBh[bbase + t * 32];
            v16bf bl = sBl[bbase + t * 32];
            acc[t] = __builtin_amdgcn_wmma_f32_16x16x32_bf16(
                false, al.v, false, bh, (short)0, acc[t], false, false);
            acc[t] = __builtin_amdgcn_wmma_f32_16x16x32_bf16(
                false, ah.v, false, bl, (short)0, acc[t], false, false);
            acc[t] = __builtin_amdgcn_wmma_f32_16x16x32_bf16(
                false, ah.v, false, bh, (short)0, acc[t], false, false);
        }
    }

    // C layout: col = l15 (+ t*16), row = m0 + half*8 + v
    float* orow = out + (m0 + (size_t)half * 8) * NCLS + l15;
    #pragma unroll
    for (int t = 0; t < 4; ++t) {
        #pragma unroll
        for (int v = 0; v < 8; ++v) {
            orow[(size_t)v * NCLS + t * 16] = acc[t][v] * INV_TAU;
        }
    }
}

extern "C" void kernel_launch(void* const* d_in, const int* in_sizes, int n_in,
                              void* d_out, int out_size, void* d_ws, size_t ws_size,
                              hipStream_t stream) {
    const float* emb = (const float*)d_in[0];
    unsigned short* Bhi = (unsigned short*)d_ws;          // 40960 ushorts (80 KB)
    unsigned short* Blo = Bhi + 40960;                    // 40960 ushorts (80 KB)
    float* out = (float*)d_out;

    proto_kernel<<<dim3(NCLS), dim3(256), 0, stream>>>(emb, Bhi, Blo);

    const float* query = emb + (size_t)NCLS * KSHOT * DDIM;   // skip 4096 support rows
    logits_kernel<<<dim3(QROWS / 128), dim3(256), 163840, stream>>>(query, Bhi, out);
}
